// MultiHeadAttention_6373731468059
// MI455X (gfx1250) — matrix operand order repaired and checked by reading, not compiled, not run on hardware
//
#include <hip/hip_runtime.h>
#include <hip/hip_bf16.h>

#define D_MODEL 1024
#define NUM_HEADS 16
#define DK 64
#define BATCH 2
#define SEQ 2048

typedef __attribute__((ext_vector_type(16))) __bf16 v16bf;
typedef __attribute__((ext_vector_type(8)))  float  v8f;

__device__ __forceinline__ v8f wmma_bf16(v16bf a, v16bf b, v8f c) {
  // 8 args: (neg_a, A, neg_b, B, c_mod, C, reuse_a, reuse_b)
  return __builtin_amdgcn_wmma_f32_16x16x32_bf16(false, a, false, b,
                                                 (short)0, c, false, false);
}

__device__ __forceinline__ float wave_max(float v) {
#pragma unroll
  for (int off = 16; off > 0; off >>= 1) v = fmaxf(v, __shfl_xor(v, off, 32));
  return v;
}
__device__ __forceinline__ float wave_sum(float v) {
#pragma unroll
  for (int off = 16; off > 0; off >>= 1) v += __shfl_xor(v, off, 32);
  return v;
}

// Core tiled GEMM body: C[m][n] = sum_k A[m][k]*W[n][k] + bias[n]
// 256 threads = 8 waves, 128x128 tile, bf16 operands double-buffered in LDS.
// Ping-pong pipeline: global loads for tile i+1 issue before the WMMA chain
// of tile i; one barrier per k-step.
__device__ __forceinline__ void gemm_body(
    const float* __restrict__ A, const float* __restrict__ W,
    const float* __restrict__ bias, float* __restrict__ C,
    int M, int N, int K, int m0, int n0)
{
  __shared__ __bf16 As[2][128][32];   // 8 KB x2
  __shared__ __bf16 Ws[2][128][32];   // 8 KB x2

  const int tid  = threadIdx.x;
  const int w    = tid >> 5;
  const int lane = tid & 31;
  const int lr   = lane & 15;
  const int hi   = lane >> 4;         // lane half (0/1)
  const int ldrow = tid >> 1;
  const int ldk   = (tid & 1) * 16;

  const float* Arow = A + (size_t)(m0 + ldrow) * K + ldk;
  const float* Wrow = W + (size_t)(n0 + ldrow) * K + ldk;

  v8f acc[8];
#pragma unroll
  for (int t = 0; t < 8; ++t) acc[t] = (v8f){};

  float ra[16], rw[16];
  // Prologue: stage k-tile 0 into buffer 0.
#pragma unroll
  for (int i = 0; i < 16; ++i) { ra[i] = Arow[i]; rw[i] = Wrow[i]; }
#pragma unroll
  for (int i = 0; i < 16; ++i) {
    As[0][ldrow][ldk + i] = (__bf16)ra[i];
    Ws[0][ldrow][ldk + i] = (__bf16)rw[i];
  }
  __syncthreads();

  int cur = 0;
  for (int kb = 0; kb < K; kb += 32) {
    const bool has_next = (kb + 32 < K);
    if (has_next) {
      // Issue next-tile global loads first; WMMA chain below hides latency.
      const float* Ap2 = Arow + kb + 32;
      const float* Wp2 = Wrow + kb + 32;
#pragma unroll
      for (int i = 0; i < 16; ++i) { ra[i] = Ap2[i]; rw[i] = Wp2[i]; }
      __builtin_prefetch(Ap2 + 32, 0, 1);   // global_prefetch_b8, tile i+2
      __builtin_prefetch(Wp2 + 32, 0, 1);
    }

    // A fragment (16x32): lane<16 -> K 0..15, lane>=16 -> 16..31
    v16bf a;
#pragma unroll
    for (int e = 0; e < 16; ++e)
      a[e] = As[cur][16 * w + lr][e + 16 * hi];

    // All 8 B fragments up front: one dscnt wait covers the whole chain.
    v16bf bb[8];
#pragma unroll
    for (int t = 0; t < 8; ++t)
#pragma unroll
      for (int e = 0; e < 16; ++e)
        bb[t][e] = Ws[cur][16 * t + lr][e + 16 * hi];

#pragma unroll
    for (int t = 0; t < 8; ++t)
      acc[t] = wmma_bf16(a, bb[t], acc[t]);

    if (has_next) {
      const int nxt = cur ^ 1;
#pragma unroll
      for (int i = 0; i < 16; ++i) {
        As[nxt][ldrow][ldk + i] = (__bf16)ra[i];
        Ws[nxt][ldrow][ldk + i] = (__bf16)rw[i];
      }
      __syncthreads();
      cur = nxt;
    }
  }

  // C/D layout: VGPR r, lane -> m = r + 8*hi, n = lane&15
#pragma unroll
  for (int t = 0; t < 8; ++t) {
    int n = n0 + 16 * t + lr;
    float bval = bias[n];
#pragma unroll
    for (int r = 0; r < 8; ++r) {
      int m = m0 + 16 * w + r + 8 * hi;
      C[(size_t)m * N + n] = acc[t][r] + bval;
    }
  }
}

// Fused Q/K/V projections: blockIdx.z selects which projection this block does.
__global__ __launch_bounds__(256) void gemm_qkv(
    const float* __restrict__ Xq, const float* __restrict__ Xk,
    const float* __restrict__ Xv,
    const float* __restrict__ Wq, const float* __restrict__ Wk,
    const float* __restrict__ Wv,
    const float* __restrict__ bq, const float* __restrict__ bk,
    const float* __restrict__ bv,
    float* __restrict__ Oq, float* __restrict__ Ok, float* __restrict__ Ov,
    int M, int N, int K)
{
  const float* A; const float* W; const float* bias; float* C;
  if (blockIdx.z == 0)      { A = Xq; W = Wq; bias = bq; C = Oq; }
  else if (blockIdx.z == 1) { A = Xk; W = Wk; bias = bk; C = Ok; }
  else                      { A = Xv; W = Wv; bias = bv; C = Ov; }
  gemm_body(A, W, bias, C, M, N, K, blockIdx.x * 128, blockIdx.y * 128);
}

__global__ __launch_bounds__(256) void gemm_xwt_bias(
    const float* __restrict__ A, const float* __restrict__ W,
    const float* __restrict__ bias, float* __restrict__ C,
    int M, int N, int K)
{
  gemm_body(A, W, bias, C, M, N, K, blockIdx.x * 128, blockIdx.y * 128);
}

// One block per (b, h, 16-row query tile). 128 threads = 4 waves.
// Phase 1: S = Q Kt (WMMA) -> LDS bf16.  Phase 2: masked softmax (wave-local).
// Phase 3: ctx = P V (WMMA). attn written to global fp32.
__global__ __launch_bounds__(128) void attn_fused(
    const float* __restrict__ Q, const float* __restrict__ Kf,
    const float* __restrict__ Vf, const int* __restrict__ mask,
    float* __restrict__ attn, float* __restrict__ ctx)
{
  __shared__ __bf16 probs[16][SEQ];   // 64 KB: scores -> exp -> normalized probs

  const int idx = blockIdx.x;
  const int qt = idx & 127;           // SEQ/16 = 128
  const int h  = (idx >> 7) & 15;
  const int b  = idx >> 11;
  const int q0 = qt * 16;

  const int tid  = threadIdx.x;
  const int w    = tid >> 5;
  const int lane = tid & 31;
  const int lr   = lane & 15;
  const int hi   = lane >> 4;

  const float* Qb = Q  + (size_t)b * SEQ * D_MODEL + (size_t)h * DK;
  const float* Kb = Kf + (size_t)b * SEQ * D_MODEL + (size_t)h * DK;
  const float* Vb = Vf + (size_t)b * SEQ * D_MODEL + (size_t)h * DK;

  // Q fragments for the block's 16 query rows (d = 0..31 and 32..63)
  v16bf aq0, aq1;
  {
    const float* qrow = Qb + (size_t)(q0 + lr) * D_MODEL;
#pragma unroll
    for (int e = 0; e < 16; ++e) {
      int d = e + 16 * hi;
      aq0[e] = (__bf16)qrow[d];
      aq1[e] = (__bf16)qrow[32 + d];
    }
  }

  // ---- Phase 1: scores. Wave w owns key columns [w*512, w*512+512). ----
  for (int t = 0; t < 32; ++t) {
    int kr0 = w * 512 + t * 16;
    const float* krow = Kb + (size_t)(kr0 + lr) * D_MODEL; // lane = key row
    v16bf b0, b1;
#pragma unroll
    for (int e = 0; e < 16; ++e) {
      b0[e] = (__bf16)krow[e + 16 * hi];        // d = 0..31
      b1[e] = (__bf16)krow[32 + e + 16 * hi];   // d = 32..63
    }
    v8f s = (v8f){};
    s = wmma_bf16(aq0, b0, s);
    s = wmma_bf16(aq1, b1, s);
#pragma unroll
    for (int r = 0; r < 8; ++r)
      probs[r + 8 * hi][kr0 + lr] = (__bf16)s[r];
  }
  __syncthreads();

  // ---- Phase 2: masked softmax. Each wave owns 4 full rows. ----
  const float scale = 0.125f;  // 1/sqrt(64)
  for (int rw = 0; rw < 4; ++rw) {
    int row = w * 4 + rw;
    const int* mrow = mask + ((size_t)b * SEQ + q0 + row) * SEQ;
    float mx = -3.0e38f;
    for (int k = lane; k < SEQ; k += 32) {
      float s = (float)probs[row][k] * scale;
      if (mrow[k] == 0) s = -1.0e9f;
      probs[row][k] = (__bf16)s;
      mx = fmaxf(mx, s);
    }
    mx = wave_max(mx);
    float sum = 0.0f;
    for (int k = lane; k < SEQ; k += 32) {
      float e_ = __expf((float)probs[row][k] - mx);
      sum += e_;
      probs[row][k] = (__bf16)e_;
    }
    sum = wave_sum(sum);
    float inv = 1.0f / sum;
    float* arow = attn + ((size_t)(b * NUM_HEADS + h) * SEQ + q0 + row) * SEQ;
    for (int k = lane; k < SEQ; k += 32) {
      float p = (float)probs[row][k] * inv;
      arow[k] = p;                     // full attention matrix output
      probs[row][k] = (__bf16)p;       // keep for PV
    }
  }
  __syncthreads();

  // ---- Phase 3: ctx = P (16x2048) @ V (2048x64). Wave w owns 16 cols. ----
  v8f acc = (v8f){};
  const int dcol0 = w * 16;
#pragma unroll 2
  for (int kt = 0; kt < SEQ / 32; ++kt) {
    int kb2 = kt * 32;
    v16bf ap, bv;
#pragma unroll
    for (int e = 0; e < 16; ++e) {
      ap[e] = probs[lr][kb2 + e + 16 * hi];
      bv[e] = (__bf16)Vb[(size_t)(kb2 + e + 16 * hi) * D_MODEL + dcol0 + lr];
    }
    acc = wmma_bf16(ap, bv, acc);
  }
#pragma unroll
  for (int r = 0; r < 8; ++r) {
    int q = q0 + r + 8 * hi;
    ctx[((size_t)b * SEQ + q) * D_MODEL + h * DK + dcol0 + lr] = acc[r];
  }
}

extern "C" void kernel_launch(void* const* d_in, const int* in_sizes, int n_in,
                              void* d_out, int out_size, void* d_ws, size_t ws_size,
                              hipStream_t stream) {
  const float* query = (const float*)d_in[0];
  const float* key   = (const float*)d_in[1];
  const float* value = (const float*)d_in[2];
  const int*   mask  = (const int*)  d_in[3];
  const float* Wq = (const float*)d_in[4];
  const float* bq = (const float*)d_in[5];
  const float* Wk = (const float*)d_in[6];
  const float* bk = (const float*)d_in[7];
  const float* Wv = (const float*)d_in[8];
  const float* bv = (const float*)d_in[9];
  const float* Wo = (const float*)d_in[10];
  const float* bo = (const float*)d_in[11];

  const size_t S = (size_t)BATCH * SEQ * D_MODEL;   // 4,194,304
  float* out  = (float*)d_out;
  float* attn = out + S;                            // tuple: (out, attn)

  float* ws  = (float*)d_ws;
  float* q   = ws;
  float* k   = ws + S;
  float* v   = ws + 2 * S;
  float* ctx = ws + 3 * S;

  const int M = BATCH * SEQ;                        // 4096

  dim3 gqkv(M / 128, D_MODEL / 128, 3);             // 32 x 8 x 3 = 768 blocks
  gemm_qkv<<<gqkv, 256, 0, stream>>>(query, key, value,
                                     Wq, Wk, Wv, bq, bk, bv,
                                     q, k, v, M, D_MODEL, D_MODEL);

  attn_fused<<<BATCH * NUM_HEADS * (SEQ / 16), 128, 0, stream>>>(q, k, v, mask, attn, ctx);

  dim3 gg(M / 128, D_MODEL / 128);                  // 32 x 8
  gemm_xwt_bias<<<gg, 256, 0, stream>>>(ctx, Wo, bo, out, M, D_MODEL, D_MODEL);
}
